// Net_90280212562085
// MI455X (gfx1250) — matrix operand order repaired and checked by reading, not compiled, not run on hardware
//
#include <hip/hip_runtime.h>
#include <cstdint>
#include <cstddef>

typedef __bf16 bf16_t;
typedef bf16_t v16bf __attribute__((ext_vector_type(16)));
typedef float  v8f   __attribute__((ext_vector_type(8)));

#define NN 32000
#define NE 64000
#define NG 2000
#define NT 8000
#define DD 64
#define EPSI 1e-5f

__device__ __forceinline__ float sigm(float x){ return 1.0f/(1.0f+__expf(-x)); }
__device__ __forceinline__ float bf2f(unsigned short u){
  unsigned v = ((unsigned)u)<<16; float f; __builtin_memcpy(&f,&v,4); return f;
}

// ---------------- WMMA fragment loads (contiguous k order, same for A and B) -------
// A: 16x32 bf16 tile, row-major A[M,K].  lane<16: row=lane, K=0..15;
// lane>=16: row=lane-16, K=16..31   (relative to k0)
__device__ __forceinline__ v16bf load_a_frag(const bf16_t* A, int ldk, int row0, int k0, int lane){
  int r = lane & 15, half = lane >> 4;
  const bf16_t* p = A + (size_t)(row0 + r) * ldk + k0 + half*16;
  v16bf a;
#pragma unroll
  for (int i=0;i<16;i++) a[i] = p[i];
  return a;
}
// B: 32x16 bf16 tile from Bt[N,K] (pre-transposed weights). lane<16: col=lane,
// K=0..15; lane>=16: col=lane-16, K=16..31 (relative to k0) => contiguous row of Bt.
__device__ __forceinline__ v16bf load_b_frag(const bf16_t* Bt, int ldk, int col0, int k0, int lane){
  int n = lane & 15, half = lane >> 4;
  const bf16_t* p = Bt + (size_t)(col0 + n) * ldk + k0 + half*16;
  v16bf b;
#pragma unroll
  for (int i=0;i<16;i++) b[i] = p[i];
  return b;
}

// ---------------- generic bf16 WMMA GEMM:  C = act(A@B + bias + addend) ----
__global__ void __launch_bounds__(256)
k_wmma_gemm(const bf16_t* __restrict__ A, const bf16_t* __restrict__ Bt,
            const float* __restrict__ bias, const float* __restrict__ addend,
            float* __restrict__ Cf, bf16_t* __restrict__ Cb,
            int M, int N, int K, int relu)
{
  int wave = blockIdx.x * (blockDim.x >> 5) + (threadIdx.x >> 5);
  int lane = threadIdx.x & 31;
  int ntiles = N >> 4;
  int mt = wave / ntiles, nt = wave % ntiles;
  if (mt * 16 >= M) return;                 // whole-wave uniform exit
  int row0 = mt << 4, col0 = nt << 4;
  v8f acc = {};
  for (int k0 = 0; k0 < K; k0 += 32){
    v16bf a = load_a_frag(A, K, row0, k0, lane);
    v16bf b = load_b_frag(Bt, K, col0, k0, lane);
    acc = __builtin_amdgcn_wmma_f32_16x16x32_bf16(false, a, false, b, (short)0, acc, false, false);
  }
  int c = col0 + (lane & 15);
  int rbase = row0 + ((lane >> 4) << 3);    // lanes<16 -> rows 0..7, else 8..15
  float bi = bias ? bias[c] : 0.0f;
#pragma unroll
  for (int j=0;j<8;j++){
    int r = rbase + j;
    float v = acc[j] + bi;
    if (addend) v += addend[(size_t)r*N + c];
    if (relu) v = fmaxf(v, 0.0f);
    if (Cf) Cf[(size_t)r*N + c] = v;
    if (Cb) Cb[(size_t)r*N + c] = (bf16_t)v;
  }
}

// ---------------- big GEMM producing W_e with fused BN affine, bf16 out ----
__global__ void __launch_bounds__(256)
k_wmma_We(const bf16_t* __restrict__ A, const bf16_t* __restrict__ Bt,
          const float* __restrict__ Ac, const float* __restrict__ Bc,
          bf16_t* __restrict__ We)
{
  const int N = 4096;
  int wave = blockIdx.x * 8 + (threadIdx.x >> 5);
  int lane = threadIdx.x & 31;
  int mt = wave >> 8, nt = wave & 255;
  if (mt >= NE/16) return;
  int row0 = mt << 4, col0 = nt << 4;
  v8f acc = {};
#pragma unroll
  for (int k0 = 0; k0 < 64; k0 += 32){
    v16bf a = load_a_frag(A, 64, row0, k0, lane);
    v16bf b = load_b_frag(Bt, 64, col0, k0, lane);
    acc = __builtin_amdgcn_wmma_f32_16x16x32_bf16(false, a, false, b, (short)0, acc, false, false);
  }
  int c = col0 + (lane & 15);
  int rbase = row0 + ((lane >> 4) << 3);
  float ac = Ac[c], bc = Bc[c];
#pragma unroll
  for (int j=0;j<8;j++)
    We[(size_t)(rbase + j)*N + c] = (bf16_t)(ac * acc[j] + bc);
}

// ---------------- helpers ---------------------------------------------------
__global__ void k_zero(float* p, size_t n){
  size_t i = (size_t)blockIdx.x*blockDim.x + threadIdx.x;
  if (i < n) p[i] = 0.0f;
}
__global__ void k_conv(const float* s, bf16_t* d, int n){
  int i = blockIdx.x*blockDim.x + threadIdx.x;
  if (i < n) d[i] = (bf16_t)s[i];
}
// W[K,N] fp32 -> Wt[N,K] bf16
__global__ void k_convT(const float* W, bf16_t* Wt, int K, int N){
  int i = blockIdx.x*blockDim.x + threadIdx.x;
  if (i >= K*N) return;
  int k = i / N, n = i % N;
  Wt[(size_t)n*K + k] = (bf16_t)W[i];
}
// small-K dense:  Y[M,N] = X[M,K] @ W[K,N] + b
__global__ void k_small_mm(const float* X, const float* W, const float* b,
                           float* Y, int M, int K, int N){
  int id = blockIdx.x*blockDim.x + threadIdx.x;
  if (id >= M*N) return;
  int m = id / N, n = id % N;
  float acc = b[n];
  for (int k=0;k<K;k++) acc += X[(size_t)m*K+k]*W[(size_t)k*N+n];
  Y[id] = acc;
}
// per-column mean/var (population) over M rows
__global__ void k_colstats(const float* X, int M, int N, float* mean, float* var){
  __shared__ float s1[256], s2[256];
  int col = blockIdx.x, t = threadIdx.x;
  float a=0.f, b=0.f;
  for (int i=t;i<M;i+=256){ float v = X[(size_t)i*N+col]; a+=v; b+=v*v; }
  s1[t]=a; s2[t]=b; __syncthreads();
  for (int off=128; off; off>>=1){ if (t<off){ s1[t]+=s1[t+off]; s2[t]+=s2[t+off]; } __syncthreads(); }
  if (t==0){ float m = s1[0]/(float)M; mean[col]=m; var[col]=s2[0]/(float)M - m*m; }
}
__global__ void k_bn_apply(const float* X, const float* mean, const float* var,
                           const float* g, const float* be,
                           float* YF, bf16_t* YB, float* YF2, bf16_t* YB2,
                           size_t MN, int N, int relu){
  size_t id = (size_t)blockIdx.x*blockDim.x + threadIdx.x;
  if (id >= MN) return;
  int c = (int)(id % (size_t)N);
  float v = g[c]*(X[id]-mean[c])*rsqrtf(var[c]+EPSI) + be[c];
  if (relu) v = fmaxf(v, 0.0f);
  if (YF) YF[id]=v; if (YB) YB[id]=(bf16_t)v;
  if (YF2) YF2[id]=v; if (YB2) YB2[id]=(bf16_t)v;
}

// Gram G += chunk(ew)^T chunk(ew)   (64x64, block handles 512 rows)
__global__ void __launch_bounds__(256) k_gram(const float* ew, float* G, int M){
  __shared__ float tile[64][64];
  int t = threadIdx.x;
  int i0 = t >> 2;                 // fixed row-index i per thread
  int j0 = (t & 3) << 4;           // 16 consecutive j per thread
  float acc[16];
#pragma unroll
  for (int q=0;q<16;q++) acc[q]=0.f;
  int r0 = blockIdx.x * 512;
  for (int rc = r0; rc < r0 + 512; rc += 64){
    __syncthreads();
    for (int i=t;i<64*64;i+=256){
      tile[i>>6][i&63] = ew[(size_t)(rc + (i>>6))*64 + (i&63)];
    }
    __syncthreads();
    for (int rr=0;rr<64;rr++){
      float xi = tile[rr][i0];
#pragma unroll
      for (int q=0;q<16;q++) acc[q] += xi * tile[rr][j0+q];
    }
  }
#pragma unroll
  for (int q=0;q<16;q++) atomicAdd(&G[i0*64 + j0 + q], acc[q]);
}

// per-output-column BN affine of edge encoder layer 2, via Gram shortcut
__global__ void __launch_bounds__(256)
k_We_affine(const float* meanE, const float* G, const float* W2,
            const float* b2, const float* g2, const float* be2,
            float* Ac, float* Bc, int M){
  __shared__ float Gs[64*64];
  __shared__ float ms[64];
  int t = threadIdx.x;
  for (int i=t;i<4096;i+=256) Gs[i]=G[i];
  if (t<64) ms[t]=meanE[t];
  __syncthreads();
  int c = blockIdx.x*256 + t;
  float w[64];
#pragma unroll
  for (int k=0;k<64;k++) w[k] = W2[(size_t)k*4096 + c];
  float m0=0.f;
#pragma unroll
  for (int k=0;k<64;k++) m0 += ms[k]*w[k];
  float q=0.f;
  for (int k=0;k<64;k++){
    float s=0.f;
#pragma unroll
    for (int l=0;l<64;l++) s += Gs[k*64+l]*w[l];
    q += w[k]*s;
  }
  float var  = q/(float)M - m0*m0;
  float mean = m0 + b2[c];
  float a = g2[c]*rsqrtf(var+EPSI);
  Ac[c]=a; Bc[c]=be2[c]-a*mean;
}

__global__ void k_deg(const int* dst, float* deg, int E){
  int e = blockIdx.x*blockDim.x + threadIdx.x;
  if (e < E) atomicAdd(&deg[dst[e]], 1.0f);
}

// one wave per edge: msg = out[src] (1x64) . W_e (64x64) bf16; scatter-add
__global__ void __launch_bounds__(256)
k_msg(const bf16_t* __restrict__ outB, const bf16_t* __restrict__ We,
      const int* __restrict__ src, const int* __restrict__ dst,
      float* __restrict__ agg, int E){
  int e = blockIdx.x*(blockDim.x>>5) + (threadIdx.x>>5);
  if (e >= E) return;
  int lane = threadIdx.x & 31;
  int s = src[e], d = dst[e];
  float x0 = (float)outB[(size_t)s*64 + lane];
  float x1 = (float)outB[(size_t)s*64 + 32 + lane];
  const bf16_t* W = We + (size_t)e*4096 + 2*lane;
  float m0=0.f, m1=0.f;
#pragma unroll 8
  for (int i=0;i<64;i++){
    if ((i & 15) == 0) __builtin_prefetch(W + (size_t)(i+16)*64, 0, 1);
    float xi = (i<32) ? __shfl(x0, i, 32) : __shfl(x1, i-32, 32);
    unsigned wp = *(const unsigned*)(const void*)(W + (size_t)i*64);
    m0 += xi * bf2f((unsigned short)(wp & 0xffffu));
    m1 += xi * bf2f((unsigned short)(wp >> 16));
  }
  atomicAdd(&agg[(size_t)d*64 + 2*lane],   m0);
  atomicAdd(&agg[(size_t)d*64 + 2*lane+1], m1);
}

__global__ void k_divdeg(float* agg, const float* deg){
  size_t id = (size_t)blockIdx.x*blockDim.x + threadIdx.x;
  if (id >= (size_t)NN*64) return;
  agg[id] /= fmaxf(deg[id>>6], 1.0f);
}

// GRU gate math + LayerNorm; one 64-thread block per node
__global__ void __launch_bounds__(64)
k_gru_ln(const float* gi, const float* gh, float* hg, bf16_t* hgB,
         float* outF, bf16_t* outB, const float* lng, const float* lnb){
  int node = blockIdx.x, d = threadIdx.x;
  size_t b3 = (size_t)node*192;
  float r = sigm(gi[b3+d]     + gh[b3+d]);
  float z = sigm(gi[b3+64+d]  + gh[b3+64+d]);
  float nn = tanhf(gi[b3+128+d] + r*gh[b3+128+d]);
  size_t b1 = (size_t)node*64 + d;
  float h = (1.0f - z)*nn + z*hg[b1];
  hg[b1]=h; hgB[b1]=(bf16_t)h;
  __shared__ float s1[64], s2[64];
  s1[d]=h; s2[d]=h*h; __syncthreads();
  for (int off=32; off; off>>=1){ if (d<off){ s1[d]+=s1[d+off]; s2[d]+=s2[d+off]; } __syncthreads(); }
  float m = s1[0]*(1.0f/64.0f), v = s2[0]*(1.0f/64.0f) - m*m;
  float o = lng[d]*(h-m)*rsqrtf(v+EPSI) + lnb[d];
  outF[b1]=o; outB[b1]=(bf16_t)o;
}

// Set2Set LSTM cell; one 256-thread block per graph
__global__ void __launch_bounds__(256)
k_lstm(const float* qstar, const float* Wih, const float* bih,
       const float* Whh, const float* bhh, float* hl, float* cl){
  int g = blockIdx.x, j = threadIdx.x;
  __shared__ float gates[256];
  __shared__ float q[128], h[64];
  if (j < 128) q[j] = qstar[(size_t)g*128 + j];
  if (j < 64)  h[j] = hl[(size_t)g*64 + j];
  __syncthreads();
  float acc = bih[j] + bhh[j];
  const float* wi = Wih + (size_t)j*128;
  for (int k=0;k<128;k++) acc += q[k]*wi[k];
  const float* wh = Whh + (size_t)j*64;
  for (int k=0;k<64;k++)  acc += h[k]*wh[k];
  gates[j] = acc; __syncthreads();
  if (j < 64){
    float ig = sigm(gates[j]),      fg = sigm(gates[64+j]);
    float gg = tanhf(gates[128+j]), og = sigm(gates[192+j]);
    float c = fg*cl[(size_t)g*64+j] + ig*gg;
    cl[(size_t)g*64+j] = c;
    hl[(size_t)g*64+j] = og*tanhf(c);
  }
}

// attention + readout: graphs own exactly 16 contiguous nodes; one wave per graph
__global__ void __launch_bounds__(32)
k_attend(const float* outF, const float* hl, float* qstar, bf16_t* qsB){
  int g = blockIdx.x, lane = threadIdx.x;
  int nbase = g*16;
  const float* hr = hl + (size_t)g*64;
  float e = -1e30f;
  if (lane < 16){
    const float* xr = outF + (size_t)(nbase+lane)*64;
    e = 0.f;
    for (int d2=0; d2<64; d2++) e += xr[d2]*hr[d2];
  }
  float emax = e;
  for (int off=8; off; off>>=1) emax = fmaxf(emax, __shfl_xor(emax, off, 32));
  float a = (lane<16) ? __expf(e - emax) : 0.0f;
  float asum = a;
  for (int off=8; off; off>>=1) asum += __shfl_xor(asum, off, 32);
  a = (lane<16) ? a/asum : 0.0f;
  float r0=0.f, r1=0.f;
  for (int n=0;n<16;n++){
    float an = __shfl(a, n, 32);
    const float* xr = outF + (size_t)(nbase+n)*64;
    r0 += an*xr[lane];
    r1 += an*xr[lane+32];
  }
  size_t qb = (size_t)g*128;
  float h0 = hr[lane], h1 = hr[lane+32];
  qstar[qb+lane]=h0;       qstar[qb+32+lane]=h1;
  qstar[qb+64+lane]=r0;    qstar[qb+96+lane]=r1;
  qsB[qb+lane]=(bf16_t)h0; qsB[qb+32+lane]=(bf16_t)h1;
  qsB[qb+64+lane]=(bf16_t)r0; qsB[qb+96+lane]=(bf16_t)r1;
}

__global__ void k_feat(const bf16_t* outB, const bf16_t* qsB, const int* ti,
                       bf16_t* feat){
  int id = blockIdx.x*blockDim.x + threadIdx.x;
  if (id >= NT*256) return;
  int t = id >> 8, j = id & 255;
  int a0 = ti[t], a1 = ti[NT + t];
  bf16_t v;
  if (j < 64)       v = outB[(size_t)a0*64 + j];
  else if (j < 128) v = outB[(size_t)a1*64 + (j-64)];
  else              v = qsB[(size_t)(a0>>4)*128 + (j-128)];
  feat[id] = v;
}

// LayerNorm(256) + ReLU; one block per row
__global__ void __launch_bounds__(256)
k_ln_relu256(const float* X, const float* g, const float* be,
             float* YF, bf16_t* YB){
  int row = blockIdx.x, t = threadIdx.x;
  size_t base = (size_t)row*256;
  float x = X[base+t];
  __shared__ float s1[256], s2[256];
  s1[t]=x; s2[t]=x*x; __syncthreads();
  for (int off=128; off; off>>=1){ if (t<off){ s1[t]+=s1[t+off]; s2[t]+=s2[t+off]; } __syncthreads(); }
  float m = s1[0]*(1.0f/256.0f), v = s2[0]*(1.0f/256.0f) - m*m;
  float o = fmaxf(g[t]*(x-m)*rsqrtf(v+EPSI)+be[t], 0.0f);
  if (YF) YF[base+t]=o; if (YB) YB[base+t]=(bf16_t)o;
}

__global__ void k_head(const float* z2, const float* W3, const float* b3,
                       const int* cls, float* out, int T){
  int t = blockIdx.x*blockDim.x + threadIdx.x;
  if (t >= T) return;
  int c = cls[t];
  float acc = b3[c];
  const float* zr = z2 + (size_t)t*256;
  for (int k=0;k<256;k++) acc += zr[k]*W3[k*8 + c];
  out[t] = acc;
}

// ---------------------------------------------------------------------------
static inline char* wsalloc(char*& p, size_t bytes){
  char* r = p; p += (bytes + 255) & ~(size_t)255; return r;
}
static void launch_gemm(hipStream_t s, const bf16_t* A, const bf16_t* Bt,
                        const float* bias, const float* add,
                        float* Cf, bf16_t* Cb, int M, int N, int K, int relu){
  int waves = (M/16)*(N/16);
  int blocks = (waves + 7)/8;
  k_wmma_gemm<<<blocks, 256, 0, s>>>(A, Bt, bias, add, Cf, Cb, M, N, K, relu);
}
static void launch_zero(hipStream_t s, float* p, size_t n){
  k_zero<<<(unsigned)((n+255)/256), 256, 0, s>>>(p, n);
}

extern "C" void kernel_launch(void* const* d_in, const int* in_sizes, int n_in,
                              void* d_out, int out_size, void* d_ws, size_t ws_size,
                              hipStream_t stream)
{
  const float* x         = (const float*)d_in[0];
  const float* edge_attr = (const float*)d_in[1];
  const float* pre_W1 = (const float*)d_in[2];  const float* pre_b1 = (const float*)d_in[3];
  const float* pre_g1 = (const float*)d_in[4];  const float* pre_be1= (const float*)d_in[5];
  const float* pre_W2 = (const float*)d_in[6];  const float* pre_b2 = (const float*)d_in[7];
  const float* pre_g2 = (const float*)d_in[8];  const float* pre_be2= (const float*)d_in[9];
  const float* enc_W1 = (const float*)d_in[10]; const float* enc_b1 = (const float*)d_in[11];
  const float* enc_g1 = (const float*)d_in[12]; const float* enc_be1= (const float*)d_in[13];
  const float* enc_W2 = (const float*)d_in[14]; const float* enc_b2 = (const float*)d_in[15];
  const float* enc_g2 = (const float*)d_in[16]; const float* enc_be2= (const float*)d_in[17];
  const float* root_W = (const float*)d_in[18]; const float* root_b = (const float*)d_in[19];
  const float* gru_Wih= (const float*)d_in[20]; const float* gru_Whh= (const float*)d_in[21];
  const float* gru_bih= (const float*)d_in[22]; const float* gru_bhh= (const float*)d_in[23];
  const float* gru_lng= (const float*)d_in[24]; const float* gru_lnb= (const float*)d_in[25];
  const float* lstm_Wih=(const float*)d_in[26]; const float* lstm_Whh=(const float*)d_in[27];
  const float* lstm_bih=(const float*)d_in[28]; const float* lstm_bhh=(const float*)d_in[29];
  const float* pr_W1  = (const float*)d_in[30]; const float* pr_b1  = (const float*)d_in[31];
  const float* pr_g1  = (const float*)d_in[32]; const float* pr_be1 = (const float*)d_in[33];
  const float* pr_W2  = (const float*)d_in[34]; const float* pr_b2  = (const float*)d_in[35];
  const float* pr_g2  = (const float*)d_in[36]; const float* pr_be2 = (const float*)d_in[37];
  const float* pr_W3  = (const float*)d_in[38]; const float* pr_b3  = (const float*)d_in[39];
  const int* edge_index   = (const int*)d_in[40];
  const int* target_index = (const int*)d_in[42];
  const int* target_class = (const int*)d_in[43];
  float* dout = (float*)d_out;
  const int* src = edge_index;
  const int* dst = edge_index + NE;

  char* p = (char*)d_ws;
  float*  h1F  = (float*) wsalloc(p, (size_t)NN*64*4);
  bf16_t* h1B  = (bf16_t*)wsalloc(p, (size_t)NN*64*2);
  float*  tmpF = (float*) wsalloc(p, (size_t)NN*64*4);
  float*  outF = (float*) wsalloc(p, (size_t)NN*64*4);
  bf16_t* outB = (bf16_t*)wsalloc(p, (size_t)NN*64*2);
  float*  hgF  = (float*) wsalloc(p, (size_t)NN*64*4);
  bf16_t* hgB  = (bf16_t*)wsalloc(p, (size_t)NN*64*2);
  bf16_t* mB   = (bf16_t*)wsalloc(p, (size_t)NN*64*2);
  float*  giF  = (float*) wsalloc(p, (size_t)NN*192*4);
  float*  ghF  = (float*) wsalloc(p, (size_t)NN*192*4);
  float*  aggF = (float*) wsalloc(p, (size_t)NN*64*4);
  float*  degF = (float*) wsalloc(p, (size_t)NN*4);
  float*  ewF  = (float*) wsalloc(p, (size_t)NE*64*4);
  bf16_t* ewB  = (bf16_t*)wsalloc(p, (size_t)NE*64*2);
  float*  G    = (float*) wsalloc(p, 4096*4);
  float*  mean1= (float*) wsalloc(p, 64*4);
  float*  var1 = (float*) wsalloc(p, 64*4);
  float*  mean2= (float*) wsalloc(p, 64*4);
  float*  var2 = (float*) wsalloc(p, 64*4);
  float*  meanE= (float*) wsalloc(p, 64*4);
  float*  varE = (float*) wsalloc(p, 64*4);
  float*  meanW= (float*) wsalloc(p, 64*4);
  float*  varW = (float*) wsalloc(p, 64*4);
  float*  Ac   = (float*) wsalloc(p, 4096*4);
  float*  Bc   = (float*) wsalloc(p, 4096*4);
  bf16_t* W2t  = (bf16_t*)wsalloc(p, (size_t)4096*64*2);
  bf16_t* preW2t=(bf16_t*)wsalloc(p, 64*64*2);
  bf16_t* rootWt=(bf16_t*)wsalloc(p, 64*64*2);
  bf16_t* WihB = (bf16_t*)wsalloc(p, 192*64*2);
  bf16_t* WhhB = (bf16_t*)wsalloc(p, 192*64*2);
  bf16_t* prW1t= (bf16_t*)wsalloc(p, 256*256*2);
  bf16_t* prW2t= (bf16_t*)wsalloc(p, 256*256*2);
  float*  qstar= (float*) wsalloc(p, (size_t)NG*128*4);
  bf16_t* qsB  = (bf16_t*)wsalloc(p, (size_t)NG*128*2);
  float*  hlF  = (float*) wsalloc(p, (size_t)NG*64*4);
  float*  clF  = (float*) wsalloc(p, (size_t)NG*64*4);
  bf16_t* featB= (bf16_t*)wsalloc(p, (size_t)NT*256*2);
  float*  z1F  = (float*) wsalloc(p, (size_t)NT*256*4);
  bf16_t* z1B  = (bf16_t*)wsalloc(p, (size_t)NT*256*2);
  float*  z2F  = (float*) wsalloc(p, (size_t)NT*256*4);
  bf16_t* WeB  = (bf16_t*)wsalloc(p, (size_t)NE*4096*2);   // 524 MB, last

  // ---- weight prep (bf16, transposed to [N,K] where needed) ----
  k_convT<<<(64*64+255)/256,   256, 0, stream>>>(pre_W2, preW2t, 64, 64);
  k_convT<<<(64*64+255)/256,   256, 0, stream>>>(root_W, rootWt, 64, 64);
  k_convT<<<(64*4096+255)/256, 256, 0, stream>>>(enc_W2, W2t, 64, 4096);
  k_convT<<<(256*256+255)/256, 256, 0, stream>>>(pr_W1, prW1t, 256, 256);
  k_convT<<<(256*256+255)/256, 256, 0, stream>>>(pr_W2, prW2t, 256, 256);
  k_conv <<<(192*64+255)/256,  256, 0, stream>>>(gru_Wih, WihB, 192*64);
  k_conv <<<(192*64+255)/256,  256, 0, stream>>>(gru_Whh, WhhB, 192*64);

  // ---- preprocess MLP ----
  k_small_mm<<<(NN*64+255)/256, 256, 0, stream>>>(x, pre_W1, pre_b1, h1F, NN, 13, 64);
  k_colstats<<<64, 256, 0, stream>>>(h1F, NN, 64, mean1, var1);
  k_bn_apply<<<(unsigned)(((size_t)NN*64+255)/256), 256, 0, stream>>>(
      h1F, mean1, var1, pre_g1, pre_be1, h1F, h1B, nullptr, nullptr, (size_t)NN*64, 64, 1);
  launch_gemm(stream, h1B, preW2t, pre_b2, nullptr, tmpF, nullptr, NN, 64, 64, 0);
  k_colstats<<<64, 256, 0, stream>>>(tmpF, NN, 64, mean2, var2);
  k_bn_apply<<<(unsigned)(((size_t)NN*64+255)/256), 256, 0, stream>>>(
      tmpF, mean2, var2, pre_g2, pre_be2, outF, outB, hgF, hgB, (size_t)NN*64, 64, 1);

  // ---- edge encoder: layer1 + BN-via-Gram for layer2, W_e in bf16 ----
  k_small_mm<<<(NE*64+255)/256, 256, 0, stream>>>(edge_attr, enc_W1, enc_b1, ewF, NE, 8, 64);
  k_colstats<<<64, 256, 0, stream>>>(ewF, NE, 64, meanE, varE);
  k_bn_apply<<<(unsigned)(((size_t)NE*64+255)/256), 256, 0, stream>>>(
      ewF, meanE, varE, enc_g1, enc_be1, ewF, ewB, nullptr, nullptr, (size_t)NE*64, 64, 1);
  k_colstats<<<64, 256, 0, stream>>>(ewF, NE, 64, meanW, varW);
  launch_zero(stream, G, 4096);
  k_gram<<<NE/512, 256, 0, stream>>>(ewF, G, NE);
  k_We_affine<<<16, 256, 0, stream>>>(meanW, G, enc_W2, enc_b2, enc_g2, enc_be2, Ac, Bc, NE);
  k_wmma_We<<<(NE/16)*256/8, 256, 0, stream>>>(ewB, W2t, Ac, Bc, WeB);

  // ---- degrees ----
  launch_zero(stream, degF, NN);
  k_deg<<<(NE+255)/256, 256, 0, stream>>>(dst, degF, NE);

  // ---- message passing + GRU, 3 steps ----
  for (int step=0; step<3; step++){
    launch_zero(stream, aggF, (size_t)NN*64);
    k_msg<<<NE/8, 256, 0, stream>>>(outB, WeB, src, dst, aggF, NE);
    k_divdeg<<<(unsigned)(((size_t)NN*64+255)/256), 256, 0, stream>>>(aggF, degF);
    launch_gemm(stream, outB, rootWt, root_b, aggF, nullptr, mB, NN, 64, 64, 1);
    launch_gemm(stream, mB,  WihB, gru_bih, nullptr, giF, nullptr, NN, 192, 64, 0);
    launch_gemm(stream, hgB, WhhB, gru_bhh, nullptr, ghF, nullptr, NN, 192, 64, 0);
    k_gru_ln<<<NN, 64, 0, stream>>>(giF, ghF, hgF, hgB, outF, outB, gru_lng, gru_lnb);
  }

  // ---- Set2Set ----
  launch_zero(stream, qstar, (size_t)NG*128);
  launch_zero(stream, hlF, (size_t)NG*64);
  launch_zero(stream, clF, (size_t)NG*64);
  for (int it=0; it<3; it++){
    k_lstm<<<NG, 256, 0, stream>>>(qstar, lstm_Wih, lstm_bih, lstm_Whh, lstm_bhh, hlF, clF);
    k_attend<<<NG, 32, 0, stream>>>(outF, hlF, qstar, qsB);
  }

  // ---- readout head ----
  k_feat<<<(NT*256+255)/256, 256, 0, stream>>>(outB, qsB, target_index, featB);
  launch_gemm(stream, featB, prW1t, pr_b1, nullptr, z1F, nullptr, NT, 256, 256, 0);
  k_ln_relu256<<<NT, 256, 0, stream>>>(z1F, pr_g1, pr_be1, nullptr, z1B);
  launch_gemm(stream, z1B, prW2t, pr_b2, nullptr, z2F, nullptr, NT, 256, 256, 0);
  k_ln_relu256<<<NT, 256, 0, stream>>>(z2F, pr_g2, pr_be2, z2F, nullptr);
  k_head<<<(NT+255)/256, 256, 0, stream>>>(z2F, pr_W3, pr_b3, target_class, dout, NT);
}
